// JunctionGNN_69939247448486
// MI455X (gfx1250) — matrix operand order repaired and checked by reading, not compiled, not run on hardware
//
#include <hip/hip_runtime.h>
#include <hip/hip_bf16.h>

typedef __attribute__((ext_vector_type(16))) _Float16 v16h;
typedef __attribute__((ext_vector_type(8)))  _Float16 v8h;
typedef __attribute__((ext_vector_type(8)))  float    v8f;

#define N_NODES 100000
#define N_EDGES 1600000
#define HDIM    64
#define NLAYERS 3
#define EPSLN   1e-5f

// ---------------------------------------------------------------------------
// Fragment layout (CDNA5 ISA 7.12.2, 16-bit WMMA, wave32)
//  A (16x32 f16): lane&15 = M; lanes 0-15 hold K 0..7,16..23; lanes 16-31 hold
//                 K 8..15,24..31  -> two contiguous 8-half chunks per lane.
//  B (32x16 f16): lane&15 = N; lanes 0-15 hold K 0..7,16..23; lanes 16-31 K 8..15,24..31.
//  C/D (16x16 f32): VGPR r: lanes 0-15 -> (M=r, N=lane); lanes 16-31 -> (M=8+r).
// ---------------------------------------------------------------------------
static __device__ __forceinline__ v16h cat8(v8h lo, v8h hi) {
  return __builtin_shufflevector(lo, hi, 0,1,2,3,4,5,6,7,8,9,10,11,12,13,14,15);
}
static __device__ __forceinline__ v16h load_a_frag(const _Float16* base, int lane) {
  const int kbase = (lane >> 4) * 8;
  v8h lo = *(const v8h*)(base + kbase);
  v8h hi = *(const v8h*)(base + kbase + 16);
  return cat8(lo, hi);
}

// ---------------------------------------------------------------------------
// Pack row-major fp32 W[K][ncols] into B-fragments; rows k >= kvalid are zero
// (used to zero-pad the K=16 edge-embed GEMM up to the 16x16x32 shape).
// out[(kt*ntiles+nt)*32 + lane][i] : K = kt*32 + (lane>>4)*8 + i + (i>=8 ? 8 : 0),
//                                    N = nt*16 + (lane&15).
// ---------------------------------------------------------------------------
__global__ void pack_b_kernel(const float* __restrict__ W, _Float16* __restrict__ out,
                              int ncols, int ktiles, int ntiles, int kvalid) {
  int tid = blockIdx.x * blockDim.x + threadIdx.x;
  int total = ktiles * ntiles * 32;
  if (tid >= total) return;
  int lane = tid & 31;
  int ft   = tid >> 5;
  int kt   = ft / ntiles, nt = ft % ntiles;
  int n    = nt * 16 + (lane & 15);
  int kgrp = (lane >> 4) * 8;
  _Float16* dst = out + (size_t)(ft * 32 + lane) * 16;
#pragma unroll
  for (int i = 0; i < 16; ++i) {
    int k = kt * 32 + kgrp + (i < 8 ? i : i + 8);
    dst[i] = (k < kvalid) ? (_Float16)W[(size_t)k * ncols + n] : (_Float16)0.0f;
  }
}

// ---------------------------------------------------------------------------
// degree / dinv / per-edge norm
// ---------------------------------------------------------------------------
__global__ void init_deg_kernel(float* __restrict__ deg) {
  int n = blockIdx.x * blockDim.x + threadIdx.x;
  if (n < N_NODES) deg[n] = 1.0f;
}
__global__ void accum_deg_kernel(const int* __restrict__ ei, float* __restrict__ deg) {
  int e = blockIdx.x * blockDim.x + threadIdx.x;   // grid exact: E % 256 == 0
  atomicAdd(&deg[ei[N_EDGES + e]], 1.0f);
}
__global__ void rsqrt_deg_kernel(float* __restrict__ deg) {
  int n = blockIdx.x * blockDim.x + threadIdx.x;
  if (n < N_NODES) deg[n] = rsqrtf(deg[n]);
}
__global__ void enorm_kernel(const int* __restrict__ ei, const float* __restrict__ dinv,
                             float* __restrict__ enorm) {
  int e = blockIdx.x * blockDim.x + threadIdx.x;   // grid exact
  enorm[e] = dinv[ei[e]] * dinv[ei[N_EDGES + e]];
}

// ---------------------------------------------------------------------------
// h = relu(x @ W_node + b) ; also write f16 mirror. Grid exact (N*64 threads).
// ---------------------------------------------------------------------------
__global__ void node_embed_kernel(const float* __restrict__ x,
                                  const float* __restrict__ Wn,
                                  const float* __restrict__ bn,
                                  float* __restrict__ h, _Float16* __restrict__ h16) {
  int tid = blockIdx.x * blockDim.x + threadIdx.x;
  int n = tid >> 6, j = tid & 63;
  float v = fmaxf(x[n * 2] * Wn[j] + x[n * 2 + 1] * Wn[64 + j] + bn[j], 0.0f);
  h[tid] = v;
  h16[tid] = (_Float16)v;
}

__global__ void zero_kernel(float* __restrict__ p) {
  p[blockIdx.x * blockDim.x + threadIdx.x] = 0.0f;   // grid exact
}

// ---------------------------------------------------------------------------
// hW = h16 @ W_conv[i]  (100k x 64 x 64) via WMMA f16->f32.
// 4 waves/block, 16 rows/wave. Uniform per-wave tail guard keeps EXEC full.
// ---------------------------------------------------------------------------
__global__ __launch_bounds__(128) void gcn_gemm_kernel(
    const _Float16* __restrict__ h16, const _Float16* __restrict__ bfrag,
    float* __restrict__ hW) {
  const int wave = threadIdx.x >> 5, lane = threadIdx.x & 31;
  const int tile = blockIdx.x * 4 + wave;
  if (tile * 16 >= N_NODES) return;                 // uniform over the wave
  const _Float16* arow = h16 + (size_t)(tile * 16 + (lane & 15)) * HDIM;
  v16h a0 = load_a_frag(arow, lane);
  v16h a1 = load_a_frag(arow + 32, lane);
  const int nloc = lane & 15, mbase = (lane >> 4) * 8;
#pragma unroll
  for (int nt = 0; nt < 4; ++nt) {
    v8f c = {};
    v16h b0 = *(const v16h*)(bfrag + (size_t)((0 * 4 + nt) * 32 + lane) * 16);
    v16h b1 = *(const v16h*)(bfrag + (size_t)((1 * 4 + nt) * 32 + lane) * 16);
    c = __builtin_amdgcn_wmma_f32_16x16x32_f16(false, a0, false, b0, (short)0, c, false, false);
    c = __builtin_amdgcn_wmma_f32_16x16x32_f16(false, a1, false, b1, (short)0, c, false, false);
#pragma unroll
    for (int r = 0; r < 8; ++r)
      hW[(size_t)(tile * 16 + mbase + r) * HDIM + nt * 16 + nloc] = c[r];
  }
}

// ---------------------------------------------------------------------------
// agg[col] += hW[row] * enorm[e]  (fp32 atomics; agg is L2-resident)
// One thread per (edge, 4-feature chunk). Grid exact: E*16 threads.
// ---------------------------------------------------------------------------
__global__ void scatter_kernel(const float* __restrict__ hW, const int* __restrict__ ei,
                               const float* __restrict__ enorm, float* __restrict__ agg) {
  int tid = blockIdx.x * blockDim.x + threadIdx.x;
  int edge = tid >> 4, c = tid & 15;
  int row = ei[edge], col = ei[N_EDGES + edge];
  float nrm = enorm[edge];
  float4 v = *(const float4*)(hW + (size_t)row * HDIM + c * 4);
  float* dst = agg + (size_t)col * HDIM + c * 4;
  atomicAdd(dst + 0, v.x * nrm);
  atomicAdd(dst + 1, v.y * nrm);
  atomicAdd(dst + 2, v.z * nrm);
  atomicAdd(dst + 3, v.w * nrm);
}

// ---------------------------------------------------------------------------
// Self-loop + bias + LayerNorm + ReLU + residual; wave32-per-node, 2 feat/lane.
// ---------------------------------------------------------------------------
__global__ __launch_bounds__(256) void post_kernel(
    const float* __restrict__ hW, const float* __restrict__ agg,
    const float* __restrict__ dinv, const float* __restrict__ bc,
    const float* __restrict__ lg, const float* __restrict__ lb,
    float* __restrict__ h, _Float16* __restrict__ h16) {
  int node = blockIdx.x * 8 + (threadIdx.x >> 5);
  if (node >= N_NODES) return;
  int lane = threadIdx.x & 31;
  size_t base = (size_t)node * HDIM + lane * 2;
  float2 hw = *(const float2*)(hW + base);
  float2 ag = *(const float2*)(agg + base);
  float di = dinv[node], dd = di * di;
  float v0 = ag.x + hw.x * dd + bc[lane * 2];
  float v1 = ag.y + hw.y * dd + bc[lane * 2 + 1];
  float s = v0 + v1;
#pragma unroll
  for (int off = 16; off >= 1; off >>= 1) s += __shfl_xor(s, off, 32);
  float mu = s * (1.0f / 64.0f);
  float d0 = v0 - mu, d1 = v1 - mu;
  float q = d0 * d0 + d1 * d1;
#pragma unroll
  for (int off = 16; off >= 1; off >>= 1) q += __shfl_xor(q, off, 32);
  float rs = rsqrtf(q * (1.0f / 64.0f) + EPSLN);
  float o0 = fmaxf(d0 * rs * lg[lane * 2]     + lb[lane * 2],     0.0f);
  float o1 = fmaxf(d1 * rs * lg[lane * 2 + 1] + lb[lane * 2 + 1], 0.0f);
  float2 hv = *(const float2*)(h + base);
  hv.x += o0; hv.y += o1;
  *(float2*)(h + base) = hv;
  h16[base]     = (_Float16)hv.x;
  h16[base + 1] = (_Float16)hv.y;
}

// ---------------------------------------------------------------------------
// Fully wave-local fused edge head (no __syncthreads): each wave owns 16 edges.
//   stage: ef[16 x 192] = [h16[row] | h16[col] | relu(edge_attr@W_edge+b)]
//          where the edge embedding itself is a zero-padded K=32 WMMA (4 ops)
//   er1 = relu(ef @ W_ep1 + b1)  : 6x4 WMMA
//   er2 = relu(er1 @ W_ep2 + b2) : 2x2 WMMA
//   s_{from,to,turn} = er2 @ W* + b*
// 128 threads = 4 waves, 64 edges/block, grid exact (E/64 = 25000).
// ---------------------------------------------------------------------------
__global__ __launch_bounds__(128) void edge_head_kernel(
    const _Float16* __restrict__ h16, const int* __restrict__ ei,
    const float* __restrict__ edge_attr,
    const _Float16* __restrict__ bpe, const float* __restrict__ b_edge,
    const _Float16* __restrict__ bp1, const float* __restrict__ b_ep1,
    const _Float16* __restrict__ bp2, const float* __restrict__ b_ep2,
    const float* __restrict__ W_from, const float* __restrict__ b_from,
    const float* __restrict__ W_to,   const float* __restrict__ b_to,
    const float* __restrict__ W_turn, const float* __restrict__ b_turn,
    float* __restrict__ out) {
  __shared__ _Float16 ef[64 * 200];   // 192 + 8 pad halves per edge row
  __shared__ _Float16 er1[64 * 72];   // 64 + 8 pad
  __shared__ float    er2[64 * 33];   // 32 + 1 pad

  const int t = threadIdx.x;
  const int wave = t >> 5, lane = t & 31;
  const int ebase = blockIdx.x * 64;
  const int nloc = lane & 15, mbase = (lane >> 4) * 8;

  // --- stage h_row / h_col for this wave's 16 edges (h16 is L2-resident) ---
  {
    int mloc = lane >> 1, part = lane & 1;          // 2 lanes per edge
    int m = wave * 16 + mloc;
    int edge = ebase + m;
    int row = ei[edge], col = ei[N_EDGES + edge];
    const uint4* sr = (const uint4*)(h16 + (size_t)row * HDIM + part * 32);
    const uint4* sc = (const uint4*)(h16 + (size_t)col * HDIM + part * 32);
    uint4* dr = (uint4*)(ef + m * 200 + part * 32);
    uint4* dc = (uint4*)(ef + m * 200 + 64 + part * 32);
#pragma unroll
    for (int j = 0; j < 4; ++j) { dr[j] = sr[j]; dc[j] = sc[j]; }
  }

  // --- edge embedding via zero-padded WMMA: [16x16] @ [16x64], K padded to 32
  {
    const float* src = edge_attr + (size_t)(ebase + wave * 16 + (lane & 15)) * 16
                                 + (lane >> 4) * 8;
    float4 f0 = *(const float4*)(src);
    float4 f1 = *(const float4*)(src + 4);
    v8h lo;
    lo[0] = (_Float16)f0.x; lo[1] = (_Float16)f0.y;
    lo[2] = (_Float16)f0.z; lo[3] = (_Float16)f0.w;
    lo[4] = (_Float16)f1.x; lo[5] = (_Float16)f1.y;
    lo[6] = (_Float16)f1.z; lo[7] = (_Float16)f1.w;
    v8h zz = (v8h)(_Float16)0.0f;
    v16h ae = cat8(lo, zz);                          // K 16..31 are zero
#pragma unroll
    for (int nt = 0; nt < 4; ++nt) {
      v8f c = {};
      v16h b = *(const v16h*)(bpe + (size_t)(nt * 32 + lane) * 16);
      c = __builtin_amdgcn_wmma_f32_16x16x32_f16(false, ae, false, b, (short)0, c, false, false);
      int n = nt * 16 + nloc;
      float bias = b_edge[n];
#pragma unroll
      for (int r = 0; r < 8; ++r)
        ef[(wave * 16 + mbase + r) * 200 + 128 + n] = (_Float16)fmaxf(c[r] + bias, 0.0f);
    }
  }
  // ef produced & consumed by the same wave; per-wave LDS ordering suffices.

  const int mrow = wave * 16 + (lane & 15);
  v16h a[6];
#pragma unroll
  for (int kt = 0; kt < 6; ++kt) a[kt] = load_a_frag(ef + mrow * 200 + kt * 32, lane);

#pragma unroll
  for (int nt = 0; nt < 4; ++nt) {
    v8f c = {};
#pragma unroll
    for (int kt = 0; kt < 6; ++kt) {
      v16h b = *(const v16h*)(bp1 + (size_t)((kt * 4 + nt) * 32 + lane) * 16);
      c = __builtin_amdgcn_wmma_f32_16x16x32_f16(false, a[kt], false, b, (short)0, c, false, false);
    }
    int n = nt * 16 + nloc;
    float bias = b_ep1[n];
#pragma unroll
    for (int r = 0; r < 8; ++r)
      er1[(wave * 16 + mbase + r) * 72 + n] = (_Float16)fmaxf(c[r] + bias, 0.0f);
  }

  v16h a20 = load_a_frag(er1 + mrow * 72, lane);
  v16h a21 = load_a_frag(er1 + mrow * 72 + 32, lane);
#pragma unroll
  for (int nt = 0; nt < 2; ++nt) {
    v8f c = {};
    v16h b0 = *(const v16h*)(bp2 + (size_t)((0 * 2 + nt) * 32 + lane) * 16);
    v16h b1 = *(const v16h*)(bp2 + (size_t)((1 * 2 + nt) * 32 + lane) * 16);
    c = __builtin_amdgcn_wmma_f32_16x16x32_f16(false, a20, false, b0, (short)0, c, false, false);
    c = __builtin_amdgcn_wmma_f32_16x16x32_f16(false, a21, false, b1, (short)0, c, false, false);
    int n = nt * 16 + nloc;
    float bias = b_ep2[n];
#pragma unroll
    for (int r = 0; r < 8; ++r)
      er2[(wave * 16 + mbase + r) * 33 + n] = fmaxf(c[r] + bias, 0.0f);
  }

  // final 3 scalar heads: 16 edges x 3 outputs = 48 dots of length 32
  for (int task = lane; task < 48; task += 32) {
    int m = task & 15;
    int o = task >> 4;
    const float* Wh = (o == 0) ? W_from : (o == 1 ? W_to : W_turn);
    float bias = (o == 0) ? b_from[0] : (o == 1 ? b_to[0] : b_turn[0]);
    const float* src = er2 + (wave * 16 + m) * 33;
    float s = bias;
#pragma unroll
    for (int k = 0; k < 32; ++k) s += src[k] * Wh[k];
    out[(size_t)o * N_EDGES + ebase + wave * 16 + m] = s;
  }
}

// ---------------------------------------------------------------------------
extern "C" void kernel_launch(void* const* d_in, const int* in_sizes, int n_in,
                              void* d_out, int out_size, void* d_ws, size_t ws_size,
                              hipStream_t stream) {
  (void)in_sizes; (void)n_in; (void)out_size; (void)ws_size;
  const float* x         = (const float*)d_in[0];
  const float* edge_attr = (const float*)d_in[1];
  const int*   ei        = (const int*)  d_in[2];
  const float* W_node    = (const float*)d_in[3];
  const float* b_node    = (const float*)d_in[4];
  const float* W_edge    = (const float*)d_in[5];
  const float* b_edge    = (const float*)d_in[6];
  const float* W_conv    = (const float*)d_in[7];
  const float* b_conv    = (const float*)d_in[8];
  const float* ln_g      = (const float*)d_in[9];
  const float* ln_b      = (const float*)d_in[10];
  const float* W_ep1     = (const float*)d_in[11];
  const float* b_ep1     = (const float*)d_in[12];
  const float* W_ep2     = (const float*)d_in[13];
  const float* b_ep2     = (const float*)d_in[14];
  const float* W_from    = (const float*)d_in[15];
  const float* b_from    = (const float*)d_in[16];
  const float* W_to      = (const float*)d_in[17];
  const float* b_to      = (const float*)d_in[18];
  const float* W_turn    = (const float*)d_in[19];
  const float* b_turn    = (const float*)d_in[20];
  float* out = (float*)d_out;

  // workspace carve (~97 MB)
  char* w = (char*)d_ws;
  size_t off = 0;
  auto carve = [&](size_t bytes) -> void* {
    off = (off + 255) & ~(size_t)255;
    void* p = w + off;
    off += bytes;
    return p;
  };
  float*     dinv  = (float*)    carve((size_t)N_NODES * 4);
  float*     enorm = (float*)    carve((size_t)N_EDGES * 4);
  float*     h     = (float*)    carve((size_t)N_NODES * HDIM * 4);
  float*     hW    = (float*)    carve((size_t)N_NODES * HDIM * 4);
  float*     agg   = (float*)    carve((size_t)N_NODES * HDIM * 4);
  _Float16*  h16   = (_Float16*) carve((size_t)N_NODES * HDIM * 2);
  _Float16*  bconv = (_Float16*) carve((size_t)NLAYERS * 2 * 4 * 32 * 16 * 2);
  _Float16*  bpk1  = (_Float16*) carve((size_t)6 * 4 * 32 * 16 * 2);
  _Float16*  bpk2  = (_Float16*) carve((size_t)2 * 2 * 32 * 16 * 2);
  _Float16*  bpke  = (_Float16*) carve((size_t)1 * 4 * 32 * 16 * 2);

  // pack weights into WMMA fragment order
  for (int i = 0; i < NLAYERS; ++i)
    pack_b_kernel<<<1, 256, 0, stream>>>(W_conv + (size_t)i * HDIM * HDIM,
                                         bconv + (size_t)i * 2 * 4 * 32 * 16,
                                         HDIM, 2, 4, 64);
  pack_b_kernel<<<3, 256, 0, stream>>>(W_ep1, bpk1, 64, 6, 4, 192);
  pack_b_kernel<<<1, 128, 0, stream>>>(W_ep2, bpk2, 32, 2, 2, 64);
  pack_b_kernel<<<1, 128, 0, stream>>>(W_edge, bpke, 64, 1, 4, 16);  // zero-padded K

  // degrees (with self-loop) -> dinv -> per-edge norms
  init_deg_kernel<<<(N_NODES + 255) / 256, 256, 0, stream>>>(dinv);
  accum_deg_kernel<<<N_EDGES / 256, 256, 0, stream>>>(ei, dinv);
  rsqrt_deg_kernel<<<(N_NODES + 255) / 256, 256, 0, stream>>>(dinv);
  enorm_kernel<<<N_EDGES / 256, 256, 0, stream>>>(ei, dinv, enorm);

  // node embedding
  node_embed_kernel<<<(N_NODES * HDIM) / 256, 256, 0, stream>>>(x, W_node, b_node, h, h16);

  // 3 GCN layers
  const int gemm_blocks = (N_NODES / 16 + 3) / 4;  // 6250 tiles, 4 waves/block
  for (int i = 0; i < NLAYERS; ++i) {
    gcn_gemm_kernel<<<gemm_blocks, 128, 0, stream>>>(
        h16, bconv + (size_t)i * 2 * 4 * 32 * 16, hW);
    zero_kernel<<<(N_NODES * HDIM) / 256, 256, 0, stream>>>(agg);
    scatter_kernel<<<(size_t)N_EDGES * 16 / 256, 256, 0, stream>>>(hW, ei, enorm, agg);
    post_kernel<<<N_NODES / 8, 256, 0, stream>>>(
        hW, agg, dinv, b_conv + (size_t)i * HDIM,
        ln_g + (size_t)i * HDIM, ln_b + (size_t)i * HDIM, h, h16);
  }

  // fused edge MLP + heads (fully wave-local, barrier-free)
  edge_head_kernel<<<N_EDGES / 64, 128, 0, stream>>>(
      h16, ei, edge_attr, bpke, b_edge, bpk1, b_ep1, bpk2, b_ep2,
      W_from, b_from, W_to, b_to, W_turn, b_turn, out);
}
